// LSTM_67078799229364
// MI455X (gfx1250) — matrix operand order repaired and checked by reading, not compiled, not run on hardware
//
#include <hip/hip_runtime.h>
#include <hip/hip_bf16.h>
#include <math.h>

// Problem constants (match reference)
#define VOCAB 50000
#define E     300
#define EP    320     // E padded to multiple of 32 for bf16 WMMA K-chunks
#define H     512
#define G4    2048    // 4*H
#define NP    3
#define B     128
#define T     256

#define NBLK_REC 32   // one workgroup per hidden j-tile (H/16)
#define LDS_PAD  8    // bf16 elements of row padding (16B -> 4-bank skew)
#define LDS_LD   (H + LDS_PAD)

typedef __attribute__((ext_vector_type(16))) __bf16 v16bf;
typedef __attribute__((ext_vector_type(8)))  float  v8f;

// ---------- helpers ----------
__device__ __forceinline__ unsigned short f2bf(float f) {
  unsigned u = __float_as_uint(f);
  u += 0x7FFFu + ((u >> 16) & 1u);          // round-to-nearest-even
  return (unsigned short)(u >> 16);
}
__device__ __forceinline__ float sigmoidf_(float x) { return 1.0f / (1.0f + __expf(-x)); }
__device__ __forceinline__ float tanh_fast(float x) {
  float e = __expf(2.0f * x);
  return (e - 1.0f) / (e + 1.0f);
}

// A-tile (16x32 bf16, MxK): lane holds row M=lane&15.
// lanes 0-15: K = kb+0..15 contiguous; lanes 16-31: K = kb+16..31 contiguous
__device__ __forceinline__ v16bf load_a_tile(const unsigned short* __restrict__ rowptr,
                                             int kb, int lane) {
  union { v16bf v; uint4 q[2]; } u;
  const unsigned short* p0 = rowptr + kb + ((lane >> 4) & 1) * 16;
  u.q[0] = *(const uint4*)(p0);
  u.q[1] = *(const uint4*)(p0 + 8);
  return u.v;
}
// B-tile (32x16 bf16, KxN): lane holds col N=lane&15 (== a row of W, since B=W^T).
// lanes 0-15: K = kb+0..15 contiguous; lanes 16-31: K = kb+16..31 contiguous
__device__ __forceinline__ v16bf load_b_tile(const unsigned short* __restrict__ rowptr,
                                             int kb, int lane) {
  union { v16bf v; uint4 q[2]; } u;
  const unsigned short* p0 = rowptr + kb + ((lane >> 4) & 1) * 16;
  u.q[0] = *(const uint4*)(p0);
  u.q[1] = *(const uint4*)(p0 + 8);
  return u.v;
}
// Same pattern but from an LDS row (base already points at row start + kb)
__device__ __forceinline__ v16bf load_b_lds(const unsigned short* base, int lane) {
  union { v16bf v; uint4 q[2]; } u;
  const unsigned short* p0 = base + ((lane >> 4) & 1) * 16;
  u.q[0] = *(const uint4*)(p0);
  u.q[1] = *(const uint4*)(p0 + 8);
  return u.v;
}
__device__ __forceinline__ v8f wmma_bf16(v16bf a, v16bf b, v8f c) {
  return __builtin_amdgcn_wmma_f32_16x16x32_bf16(false, a, false, b, (short)0, c, false, false);
}

// Device-wide sense barrier (persistent-kernel grid sync)
__device__ __forceinline__ void grid_sync(unsigned* cnt, unsigned* gen, unsigned nb) {
  __threadfence();
  __syncthreads();
  if (threadIdx.x == 0) {
    unsigned g = __hip_atomic_load(gen, __ATOMIC_ACQUIRE, __HIP_MEMORY_SCOPE_AGENT);
    unsigned a = __hip_atomic_fetch_add(cnt, 1u, __ATOMIC_ACQ_REL, __HIP_MEMORY_SCOPE_AGENT);
    if (a == nb - 1u) {
      __hip_atomic_store(cnt, 0u, __ATOMIC_RELAXED, __HIP_MEMORY_SCOPE_AGENT);
      __hip_atomic_fetch_add(gen, 1u, __ATOMIC_RELEASE, __HIP_MEMORY_SCOPE_AGENT);
    } else {
      while (__hip_atomic_load(gen, __ATOMIC_ACQUIRE, __HIP_MEMORY_SCOPE_AGENT) == g)
        __builtin_amdgcn_s_sleep(2);
    }
  }
  __syncthreads();
}

// ---------- prep kernels ----------
__global__ void cvt_pad_kernel(const float* __restrict__ src, unsigned short* __restrict__ dst,
                               int rows, int scols, int dcols) {
  long total = (long)rows * dcols;
  for (long i = (long)blockIdx.x * blockDim.x + threadIdx.x; i < total;
       i += (long)gridDim.x * blockDim.x) {
    int r = (int)(i / dcols), c = (int)(i % dcols);
    float v = (c < scols) ? src[(long)r * scols + c] : 0.0f;
    dst[i] = f2bf(v);
  }
}

__global__ void init_state_kernel(float* hf, float* cf, unsigned short* hb0, unsigned* bar) {
  int i = blockIdx.x * blockDim.x + threadIdx.x;
  if (i < B * H) { hf[i] = 0.0f; cf[i] = 0.0f; hb0[i] = 0; }
  if (i < 2) bar[i] = 0u;
}

__global__ __launch_bounds__(256) void lens_kernel(const int* __restrict__ text,
                                                   int* __restrict__ lens) {
  __shared__ int s[256];
  int b = blockIdx.x, t = threadIdx.x;
  s[t] = (text[b * T + t] != 0) ? 1 : 0;
  __syncthreads();
  for (int off = 128; off > 0; off >>= 1) {
    if (t < off) s[t] += s[t + off];
    __syncthreads();
  }
  if (t == 0) lens[b] = s[0];
}

// ---------- phase 1: fused gather + pre-gate GEMM ----------
// xg[b,t,:] = emb[text[b,t]] @ W_ih^T + (b_ih + b_hh), M = B*T, N = 4H, K = EP
__global__ __launch_bounds__(256) void xg_gemm_kernel(
    const int* __restrict__ text, const unsigned short* __restrict__ embB,
    const unsigned short* __restrict__ wihB, const float* __restrict__ bih,
    const float* __restrict__ bhh, float* __restrict__ xg) {
  const int lane  = threadIdx.x & 31;
  const int wave  = threadIdx.x >> 5;
  const int wid   = blockIdx.x * 8 + wave;   // 0..32767
  const int mtile = wid >> 4;                // 0..2047 (M/16)
  const int ntb   = (wid & 15) * 8;          // base of 8 n-tiles
  const int ln    = lane & 15;
  const int hi    = (lane >> 4) & 1;

  const int m     = mtile * 16 + ln;         // flat (b*T + t) row
  const int token = text[m];
  const unsigned short* arow = embB + (size_t)token * EP;

  v8f acc[8];
#pragma unroll
  for (int j = 0; j < 8; ++j) {
    int n = (ntb + j) * 16 + ln;
    float bias = bih[n] + bhh[n];
#pragma unroll
    for (int r = 0; r < 8; ++r) acc[j][r] = bias;
  }

  for (int kc = 0; kc < EP / 32; ++kc) {
    const int kb = kc * 32;
    v16bf a = load_a_tile(arow, kb, lane);
    if (kc + 1 < EP / 32) __builtin_prefetch(arow + kb + 32, 0, 1);
#pragma unroll
    for (int j = 0; j < 8; ++j) {
      int n = (ntb + j) * 16 + ln;
      v16bf b = load_b_tile(wihB + (size_t)n * EP, kb, lane);
      acc[j] = wmma_bf16(a, b, acc[j]);
    }
  }

#pragma unroll
  for (int j = 0; j < 8; ++j) {
    int n = (ntb + j) * 16 + ln;
#pragma unroll
    for (int r = 0; r < 8; ++r) {
      int mr = mtile * 16 + r + 8 * hi;
      xg[(size_t)mr * G4 + n] = acc[j][r];
    }
  }
}

// ---------- phase 2: persistent LSTM recurrence ----------
// Block = one hidden j-tile (16 cols), wave = one batch m-tile, 4 accumulators =
// the 4 gate groups. Gates stay in registers; h is double-buffered in bf16 so a
// single grid sync per step suffices. W_hh slice lives in LDS for all 256 steps.
__global__ __launch_bounds__(256) void lstm_recurrent_kernel(
    const float* __restrict__ xg, const unsigned short* __restrict__ whhB,
    const int* __restrict__ lens, float* __restrict__ hf, float* __restrict__ cf,
    unsigned short* __restrict__ hb0, unsigned short* __restrict__ hb1,
    unsigned* __restrict__ bar) {
  __shared__ unsigned short ldsB[4 * 16 * LDS_LD];   // 66.6 KB of 320 KB/WGP

  const int lane  = threadIdx.x & 31;
  const int wave  = threadIdx.x >> 5;
  const int ln    = lane & 15;
  const int hi    = (lane >> 4) & 1;
  const int jt    = blockIdx.x;     // 0..31 hidden j-tile
  const int mtile = wave;           // 0..7 batch m-tile
  unsigned* cnt = bar;
  unsigned* gen = bar + 1;

  // Stage W_hh^T slice: rows n = gg*H + jt*16 + rr (gg=0..3, rr=0..15), 512 K each.
  for (int i = threadIdx.x; i < 4 * 16 * (H / 8); i += 256) {
    int row = i / (H / 8);          // 0..63 = gg*16 + rr
    int c8  = i % (H / 8);
    int gg = row >> 4, rr = row & 15;
    const unsigned short* src = whhB + (size_t)(gg * H + jt * 16 + rr) * H + c8 * 8;
    *(uint4*)(ldsB + (size_t)row * LDS_LD + c8 * 8) = *(const uint4*)src;
  }
  __syncthreads();

  const int j = jt * 16 + ln;       // hidden index this lane owns

  for (int t = 0; t < T; ++t) {
    const unsigned short* rb = (t & 1) ? hb1 : hb0;   // h read buffer
    unsigned short*       wb = (t & 1) ? hb0 : hb1;   // h write buffer

    // init accumulators from xg slice (biases already folded in)
    v8f acc[4];
#pragma unroll
    for (int gg = 0; gg < 4; ++gg) {
      int n = gg * H + j;
#pragma unroll
      for (int r = 0; r < 8; ++r) {
        int b = mtile * 16 + r + 8 * hi;
        acc[gg][r] = xg[((size_t)b * T + t) * G4 + n];
      }
    }

    // gates += h @ W_hh^T : A from global (L2), B from LDS
    const unsigned short* arow = rb + (size_t)(mtile * 16 + ln) * H;
#pragma unroll
    for (int kc = 0; kc < H / 32; ++kc) {
      const int kb = kc * 32;
      v16bf a = load_a_tile(arow, kb, lane);
#pragma unroll
      for (int gg = 0; gg < 4; ++gg) {
        const unsigned short* bbase = ldsB + (size_t)(gg * 16 + ln) * LDS_LD + kb;
        acc[gg] = wmma_bf16(a, load_b_lds(bbase, lane), acc[gg]);
      }
    }

    // LSTM cell math entirely in accumulator registers
#pragma unroll
    for (int r = 0; r < 8; ++r) {
      int b = mtile * 16 + r + 8 * hi;
      size_t e = (size_t)b * H + j;
      float i_ = sigmoidf_(acc[0][r]);
      float f_ = sigmoidf_(acc[1][r]);
      float g_ = tanh_fast(acc[2][r]);
      float o_ = sigmoidf_(acc[3][r]);
      float cn = f_ * cf[e] + i_ * g_;
      float hn = o_ * tanh_fast(cn);
      bool m = t < lens[b];
      if (m) { cf[e] = cn; hf[e] = hn; }
      wb[e] = m ? f2bf(hn) : rb[e];   // carry old h forward past sequence end
    }

    grid_sync(cnt, gen, NBLK_REC);    // publish h for next step's GEMM
  }
}

// ---------- phase 3: out = h_n @ Wd^T + bd ----------
__global__ void final_proj_kernel(const float* __restrict__ hf, const float* __restrict__ wd,
                                  const float* __restrict__ bd, float* __restrict__ out) {
  int idx = blockIdx.x * blockDim.x + threadIdx.x;
  if (idx >= B * NP) return;
  int b = idx / NP, p = idx % NP;
  float s = bd[p];
  for (int k = 0; k < H; ++k) s += hf[b * H + k] * wd[p * H + k];
  out[idx] = s;
}

// ---------- host ----------
extern "C" void kernel_launch(void* const* d_in, const int* in_sizes, int n_in,
                              void* d_out, int out_size, void* d_ws, size_t ws_size,
                              hipStream_t stream) {
  const int*   text = (const int*)d_in[0];
  const float* emb  = (const float*)d_in[1];
  const float* wih  = (const float*)d_in[2];
  const float* whh  = (const float*)d_in[3];
  const float* bih  = (const float*)d_in[4];
  const float* bhh  = (const float*)d_in[5];
  const float* wd   = (const float*)d_in[6];
  const float* bd   = (const float*)d_in[7];
  float* out = (float*)d_out;

  char* ws = (char*)d_ws;
  size_t off = 0;
  auto carve = [&](size_t bytes) -> char* {
    char* p = ws + off;
    off += (bytes + 255) & ~(size_t)255;
    return p;
  };
  float*          xg   = (float*)         carve((size_t)B * T * G4 * 4);  // 256 MB
  float*          hf   = (float*)         carve((size_t)B * H * 4);
  float*          cf   = (float*)         carve((size_t)B * H * 4);
  unsigned short* hb0  = (unsigned short*)carve((size_t)B * H * 2);
  unsigned short* hb1  = (unsigned short*)carve((size_t)B * H * 2);
  int*            lens = (int*)           carve((size_t)B * 4);
  unsigned short* embB = (unsigned short*)carve((size_t)VOCAB * EP * 2);  // 32 MB
  unsigned short* wihB = (unsigned short*)carve((size_t)G4 * EP * 2);
  unsigned short* whhB = (unsigned short*)carve((size_t)G4 * H * 2);
  unsigned*       bar  = (unsigned*)      carve(64);

  // prep: bf16 conversions (+ K padding), state init, sequence lengths
  cvt_pad_kernel<<<4096, 256, 0, stream>>>(emb, embB, VOCAB, E, EP);
  cvt_pad_kernel<<<1024, 256, 0, stream>>>(wih, wihB, G4, E, EP);
  cvt_pad_kernel<<<1024, 256, 0, stream>>>(whh, whhB, G4, H, H);
  init_state_kernel<<<256, 256, 0, stream>>>(hf, cf, hb0, bar);
  lens_kernel<<<B, 256, 0, stream>>>(text, lens);

  // pre-gate GEMM (gather fused)
  xg_gemm_kernel<<<4096, 256, 0, stream>>>(text, embB, wihB, bih, bhh, xg);

  // persistent recurrence: in-register gates, one grid sync per step
  lstm_recurrent_kernel<<<NBLK_REC, 256, 0, stream>>>(xg, whhB, lens, hf, cf, hb0, hb1, bar);

  // output projection
  final_proj_kernel<<<2, 256, 0, stream>>>(hf, wd, bd, out);
}
